// NeighborAttention_15882789060650
// MI455X (gfx1250) — matrix operand order repaired and checked by reading, not compiled, not run on hardware
//
#include <hip/hip_runtime.h>
#include <hip/hip_bf16.h>

// ---------------------------------------------------------------------------
// NeighborAttention for MI455X (gfx1250, wave32).
//
// Roofline: ~65 GFLOP vs ~0.52 GB traffic (h_E dominates at 492 MB -> ~21 us
// at 23.3 TB/s). bf16 WMMA (v_wmma_f32_16x16x32_bf16, f32 accumulate) keeps
// the dominant K/V projections near the memory roof; fp32 WMMA (16x16x4)
// would be ~8x more matrix instructions. Small GEMV-like stages (Q, W_O,
// softmax) stay f32 VALU for accuracy.
//
// One node per 128-thread block (4 wave32 waves, wave == head).
// Weights W_K/W_V are pre-converted to bf16 once into d_ws by a prep kernel
// so the 30000 main blocks load B-fragments as raw global_load_b128 with no
// per-block f32->bf16 VALU conversion (deterministic fallback converts
// in-kernel if ws_size is too small).
// ---------------------------------------------------------------------------

typedef __bf16 bf16_t;
typedef bf16_t v16bf __attribute__((ext_vector_type(16)));
typedef float  v8f   __attribute__((ext_vector_type(8)));

#define E_STR  136   // bf16 elements per LDS row of E (32x128 tile), padded
#define KV_STR 132   // f32 elements per LDS row of K/V (32x128), padded

union AFrag { v16bf v; uint4  q[2];   };
union BFrag { v16bf v; bf16_t e[16];  };
union CFrag { v8f   v; float  f[8];   };

// A-matrix (16x32 bf16) lane layout used here (contiguous k, same order as B):
//   lane l: M = mBase + (l&15); lanes<16 hold K={0..15}, lanes>=16 K={16..31}
//   -> two contiguous 8-bf16 (16B) runs per lane: ds_load_b128 x2.
__device__ __forceinline__ v16bf load_a_frag(const bf16_t* e_sh, int mBase, int kBase, int lane) {
    const int m   = mBase + (lane & 15);
    const int kk0 = (lane < 16) ? 0 : 16;
    AFrag a;
    a.q[0] = *(const uint4*)(e_sh + m * E_STR + kBase + kk0);
    a.q[1] = *(const uint4*)(e_sh + m * E_STR + kBase + kk0 + 8);
    return a.v;
}

// B-matrix (32x16 bf16, B[k][j] = W[j][k]) lane layout:
//   lane l: N = nBase + (l&15); K = kBase + (l>>4)*16 + {0..15} contiguous
//   -> 16 contiguous bf16 of one weight row: global_load_b128 x2 (fast path).
__device__ __forceinline__ v16bf load_b_frag_bf16(const bf16_t* Wb, int nBase, int kBase, int lane) {
    const int j  = nBase + (lane & 15);
    const int c0 = kBase + ((lane >> 4) << 4);
    const bf16_t* p = Wb + j * 128 + c0;         // 32-byte aligned (c0 % 16 == 0)
    AFrag b;
    b.q[0] = *(const uint4*)(p);
    b.q[1] = *(const uint4*)(p + 8);
    return b.v;
}

// Fallback: same fragment converted from the original f32 weights in-kernel.
__device__ __forceinline__ v16bf load_b_frag_f32(const float* W, int nBase, int kBase, int lane) {
    const int j  = nBase + (lane & 15);
    const int c0 = kBase + ((lane >> 4) << 4);
    const float4* p = (const float4*)(W + j * 128 + c0);
    float4 w0 = p[0], w1 = p[1], w2 = p[2], w3 = p[3];
    BFrag b;
    b.e[ 0] = (bf16_t)w0.x; b.e[ 1] = (bf16_t)w0.y; b.e[ 2] = (bf16_t)w0.z; b.e[ 3] = (bf16_t)w0.w;
    b.e[ 4] = (bf16_t)w1.x; b.e[ 5] = (bf16_t)w1.y; b.e[ 6] = (bf16_t)w1.z; b.e[ 7] = (bf16_t)w1.w;
    b.e[ 8] = (bf16_t)w2.x; b.e[ 9] = (bf16_t)w2.y; b.e[10] = (bf16_t)w2.z; b.e[11] = (bf16_t)w2.w;
    b.e[12] = (bf16_t)w3.x; b.e[13] = (bf16_t)w3.y; b.e[14] = (bf16_t)w3.z; b.e[15] = (bf16_t)w3.w;
    return b.v;
}

// C/D (16x16 f32) lane layout: VGPR r -> row r (lanes<16) / r+8 (lanes>=16), col = lane&15.
__device__ __forceinline__ void store_c_frag(float* dst, int mBase, int nBase, int lane, v8f c) {
    const int col   = nBase + (lane & 15);
    const int rbase = mBase + ((lane < 16) ? 0 : 8);
    CFrag cf; cf.v = c;
#pragma unroll
    for (int r = 0; r < 8; ++r)
        dst[(rbase + r) * KV_STR + col] = cf.f[r];
}

// ---- Prep: convert W_K / W_V (f32 [128][128]) -> bf16 into d_ws ----------
__global__ void __launch_bounds__(256)
convert_weights_kernel(const float* __restrict__ WK, const float* __restrict__ WV,
                       bf16_t* __restrict__ ws) {
    const int i = blockIdx.x * 256 + threadIdx.x;       // 0 .. 16383
    ws[i]         = (bf16_t)WK[i];
    ws[16384 + i] = (bf16_t)WV[i];
}

template <bool PREBF16>
__global__ void __launch_bounds__(128)
neighbor_attention_kernel(const float* __restrict__ hV,   // [N,128]
                          const float* __restrict__ hE,   // [N,32,128]
                          const int*   __restrict__ mask, // [N,32]
                          const float* __restrict__ WQ,   // [128,128]
                          const void*  __restrict__ WKp,  // bf16 or f32 per PREBF16
                          const void*  __restrict__ WVp,
                          const float* __restrict__ WO,
                          float* __restrict__ out)        // [N,128]
{
    __shared__ __align__(16) bf16_t e_sh[32 * E_STR];   // h_E[n] in bf16,  8.5 KB
    __shared__ __align__(16) float  k_sh[32 * KV_STR];  // K proj (f32),   16.5 KB
    __shared__ __align__(16) float  v_sh[32 * KV_STR];  // V proj (f32),   16.5 KB
    __shared__ float hv_sh[128];
    __shared__ float q_sh[128];
    __shared__ float o_sh[128];

    const int n    = blockIdx.x;
    const int t    = threadIdx.x;
    const int lane = t & 31;
    const int head = t >> 5;            // 4 waves -> 4 heads

    // ---- Phase 1: stage h_V and h_E (f32 -> bf16) into LDS -------------
    hv_sh[t] = hV[(size_t)n * 128 + t];
    const float4* E4 = (const float4*)(hE + (size_t)n * 32 * 128);
#pragma unroll
    for (int i = 0; i < 8; ++i) {
        const int f   = t + i * 128;    // float4 index within 32x128 tile
        const int row = f >> 5;
        const int c4  = f & 31;
        float4 e = E4[f];
        union { unsigned int u[2]; bf16_t b[4]; } pk;
        pk.b[0] = (bf16_t)e.x; pk.b[1] = (bf16_t)e.y;
        pk.b[2] = (bf16_t)e.z; pk.b[3] = (bf16_t)e.w;
        unsigned int* dst = (unsigned int*)(e_sh + row * E_STR + c4 * 4);
        dst[0] = pk.u[0]; dst[1] = pk.u[1];
    }
    __syncthreads();

    // ---- Phase 2a: Q = h_V @ WQ^T (f32 VALU, one feature per thread) ---
    {
        float acc = 0.f;
        const float4* W4 = (const float4*)(WQ + t * 128);
#pragma unroll
        for (int c4 = 0; c4 < 32; ++c4) {
            float4 w = W4[c4];
            acc += hv_sh[c4 * 4 + 0] * w.x + hv_sh[c4 * 4 + 1] * w.y
                 + hv_sh[c4 * 4 + 2] * w.z + hv_sh[c4 * 4 + 3] * w.w;
        }
        q_sh[t] = acc;
    }

    // ---- Phase 2b: K/V projections via bf16 WMMA (per-head columns) ----
    // Per wave: 2 proj x 4 ksteps x (2 Mtiles x 2 Ntiles) = 32 v_wmma.
    const int nb = head * 32;
#pragma unroll
    for (int proj = 0; proj < 2; ++proj) {
        const void* W = proj ? WVp : WKp;
        float* dst    = proj ? v_sh : k_sh;
        v8f c00 = {}, c01 = {}, c10 = {}, c11 = {};
#pragma unroll
        for (int ks = 0; ks < 4; ++ks) {
            const int kb = ks * 32;
            v16bf a0 = load_a_frag(e_sh, 0,  kb, lane);
            v16bf a1 = load_a_frag(e_sh, 16, kb, lane);
            v16bf b0, b1;
            if (PREBF16) {
                b0 = load_b_frag_bf16((const bf16_t*)W, nb,      kb, lane);
                b1 = load_b_frag_bf16((const bf16_t*)W, nb + 16, kb, lane);
            } else {
                b0 = load_b_frag_f32((const float*)W, nb,      kb, lane);
                b1 = load_b_frag_f32((const float*)W, nb + 16, kb, lane);
            }
            c00 = __builtin_amdgcn_wmma_f32_16x16x32_bf16(false, a0, false, b0, (short)0, c00, false, false);
            c01 = __builtin_amdgcn_wmma_f32_16x16x32_bf16(false, a0, false, b1, (short)0, c01, false, false);
            c10 = __builtin_amdgcn_wmma_f32_16x16x32_bf16(false, a1, false, b0, (short)0, c10, false, false);
            c11 = __builtin_amdgcn_wmma_f32_16x16x32_bf16(false, a1, false, b1, (short)0, c11, false, false);
        }
        store_c_frag(dst, 0,  nb,      lane, c00);
        store_c_frag(dst, 0,  nb + 16, lane, c01);
        store_c_frag(dst, 16, nb,      lane, c10);
        store_c_frag(dst, 16, nb + 16, lane, c11);
    }
    __syncthreads();

    // ---- Phase 3: masked softmax attention (per wave = per head) -------
    {
        const int base = head * 32;
        // lane = neighbor k: logit = <Q_head, K_k_head> / sqrt(32)
        float logit = 0.f;
#pragma unroll
        for (int d = 0; d < 32; ++d)
            logit += q_sh[base + d] * k_sh[lane * KV_STR + base + d];
        logit *= 0.17677669529663687f;                 // 1/sqrt(32)

        const int   mk      = mask[(size_t)n * 32 + lane];
        const float NEG_INF = -3.402823466e38f;
        logit = mk ? logit : NEG_INF;

        float mx = logit;
#pragma unroll
        for (int off = 16; off >= 1; off >>= 1)
            mx = fmaxf(mx, __shfl_xor(mx, off, 32));
        float ex = __expf(logit - mx);
        float sm = ex;
#pragma unroll
        for (int off = 16; off >= 1; off >>= 1)
            sm += __shfl_xor(sm, off, 32);
        float attend = mk ? (ex / sm) : 0.f;           // softmax * mask

        // lane = feature d: out[d] = sum_k attend[k] * V[k][d]
        float acc = 0.f;
#pragma unroll
        for (int k2 = 0; k2 < 32; ++k2) {
            float a = __shfl(attend, k2, 32);
            acc += a * v_sh[k2 * KV_STR + base + lane];
        }
        o_sh[base + lane] = acc;
    }
    __syncthreads();

    // ---- Phase 4: out @ WO^T (f32 VALU, one feature per thread) --------
    {
        float acc = 0.f;
        const float4* W4 = (const float4*)(WO + t * 128);
#pragma unroll
        for (int c4 = 0; c4 < 32; ++c4) {
            float4 w = W4[c4];
            acc += o_sh[c4 * 4 + 0] * w.x + o_sh[c4 * 4 + 1] * w.y
                 + o_sh[c4 * 4 + 2] * w.z + o_sh[c4 * 4 + 3] * w.w;
        }
        out[(size_t)n * 128 + t] = acc;
    }
}

extern "C" void kernel_launch(void* const* d_in, const int* in_sizes, int n_in,
                              void* d_out, int out_size, void* d_ws, size_t ws_size,
                              hipStream_t stream) {
    const float* hV   = (const float*)d_in[0];   // [N,128]
    const float* hE   = (const float*)d_in[1];   // [N,32,128]
    const int*   mask = (const int*)  d_in[2];   // [N,32]
    const float* WQ   = (const float*)d_in[3];
    const float* WK   = (const float*)d_in[4];
    const float* WV   = (const float*)d_in[5];
    const float* WO   = (const float*)d_in[6];
    float* out = (float*)d_out;

    const int    N         = in_sizes[0] / 128;            // 30000
    const size_t ws_needed = 2u * 128u * 128u * sizeof(bf16_t);  // 64 KB

    if (ws_size >= ws_needed) {
        // Fast path: hoist the f32->bf16 weight conversion out of the 30000
        // main blocks (deterministic: runs identically every call).
        bf16_t* wb = (bf16_t*)d_ws;
        convert_weights_kernel<<<64, 256, 0, stream>>>(WK, WV, wb);
        neighbor_attention_kernel<true><<<N, 128, 0, stream>>>(
            hV, hE, mask, WQ, (const void*)wb, (const void*)(wb + 16384), WO, out);
    } else {
        neighbor_attention_kernel<false><<<N, 128, 0, stream>>>(
            hV, hE, mask, WQ, (const void*)WK, (const void*)WV, WO, out);
    }
}
